// AdaptiveLTCCell_43542378447325
// MI455X (gfx1250) — matrix operand order repaired and checked by reading, not compiled, not run on hardware
//
#include <hip/hip_runtime.h>

// ---------------------------------------------------------------------------
// AdaptiveLTCCell on MI455X (gfx1250, wave32, WMMA).
// B = D_IN = S = 1024, TIME_LIMIT = 16, ODE_UNFOLDS = 3.
//   * X = inputs@Wx + b hoisted (one GEMM).
//   * 45 dependent h@Wh GEMMs, split-bf16 (hi/lo) WMMA, fp32 accumulate.
//     h lives as a split-bf16 pair produced by the previous epilogue ->
//     zero conversion work in the K-loop; staging is pure block copies via
//     gfx1250 async global->LDS, double buffered (ASYNCcnt pipelined).
//   * LTC ODE update fused in the GEMM epilogue; halting GEMV + ACT fused.
// ---------------------------------------------------------------------------

#define DIM   1024
#define NSTEP 15
#define NUNF  3
#define DT    (1.0f / 3.0f)
#define BUDGET 0.99f
#define TIME_PENALTY 0.01f

typedef __attribute__((ext_vector_type(16))) __bf16 v16bf;
typedef __attribute__((ext_vector_type(8)))  float  v8f;
typedef int v4i __attribute__((vector_size(16)));     // matches builtin proto

#define AS1 __attribute__((address_space(1)))
#define AS3 __attribute__((address_space(3)))

#if defined(__has_builtin)
#if __has_builtin(__builtin_amdgcn_global_load_async_to_lds_b128)
#define HAVE_ASYNC_LDS 1
#endif
#endif
#ifndef HAVE_ASYNC_LDS
#define HAVE_ASYNC_LDS 0
#warning "gfx1250 global_load_async_to_lds_b128 builtin NOT available; using sync LDS staging"
#endif

#if HAVE_ASYNC_LDS
#if defined(__has_builtin) && __has_builtin(__builtin_amdgcn_s_wait_asynccnt)
#define ASYNC_WAIT(imm) __builtin_amdgcn_s_wait_asynccnt(imm)
#else
#define ASYNC_WAIT(imm) asm volatile("s_wait_asynccnt %0" ::"i"(imm) : "memory")
#endif
#define STAGE16(src, dst)                                                   \
    __builtin_amdgcn_global_load_async_to_lds_b128(                         \
        (AS1 v4i*)(uintptr_t)(const void*)(src),                            \
        (AS3 v4i*)(unsigned int)(uintptr_t)(void*)(dst), 0, 0)
#define AWAIT1() ASYNC_WAIT(1)
#define AWAIT0() ASYNC_WAIT(0)
#else
#define STAGE16(src, dst)                                                   \
    (*reinterpret_cast<uint4*>(dst) = *reinterpret_cast<const uint4*>(src))
#define AWAIT1()
#define AWAIT0()
#endif

union BF16Frag { v16bf v; uint4 u[2]; };

__device__ __forceinline__ void split_bf16(float x, __bf16& hi, __bf16& lo) {
    hi = (__bf16)x;
    lo = (__bf16)(x - (float)hi);           // residual exact in f32
}

#define WMMA_BF16(A, B, C)                                                  \
    __builtin_amdgcn_wmma_f32_16x16x32_bf16(false, (A), false, (B),         \
                                            (short)0, (C), false, false)

// ---------------------------------------------------------------------------
// Transpose + split-convert f32 [K][N] weights -> bf16 hi/lo [N][K].
// ---------------------------------------------------------------------------
__global__ __launch_bounds__(256) void transpose_split_bf16(
    const float* __restrict__ W, __bf16* __restrict__ WT_hi, __bf16* __restrict__ WT_lo)
{
    __shared__ float t[32][33];
    const int tx = threadIdx.x, ty = threadIdx.y;        // 32 x 8
    const int bx = blockIdx.x * 32, by = blockIdx.y * 32;
#pragma unroll
    for (int j = 0; j < 32; j += 8)
        t[ty + j][tx] = W[(size_t)(by + ty + j) * DIM + bx + tx];
    __syncthreads();
#pragma unroll
    for (int j = 0; j < 32; j += 8) {
        __bf16 hi, lo;
        split_bf16(t[tx][ty + j], hi, lo);
        const size_t o = (size_t)(bx + ty + j) * DIM + by + tx;
        WT_hi[o] = hi;
        WT_lo[o] = lo;
    }
}

// Elementwise f32 -> split bf16 (for `inputs` and `hidden0`).
__global__ __launch_bounds__(256) void convert_split(
    const float* __restrict__ in, __bf16* __restrict__ hi, __bf16* __restrict__ lo)
{
    const size_t i = (size_t)blockIdx.x * 256 + threadIdx.x;
    __bf16 h, l;
    split_bf16(in[i], h, l);
    hi[i] = h;
    lo[i] = l;
}

// ---------------------------------------------------------------------------
// One K-step (K=32) of the split-bf16 product for both 16-row tiles.
// A fragments from LDS; B fragments from global (L2-resident weights).
// hi*hi + hi*lo + lo*hi, f32 accumulate; acc0/acc1 chained directly.
// ---------------------------------------------------------------------------
__device__ __forceinline__ void mma_step(
    const __bf16 (*__restrict__ tHi)[32], const __bf16 (*__restrict__ tLo)[32],
    const __bf16* __restrict__ BTHi, const __bf16* __restrict__ BTLo,
    size_t boff, int lm, int half, v8f& acc0, v8f& acc1)
{
    BF16Frag bh, bl, ah0, al0, ah1, al1;
    bh.u[0]  = *reinterpret_cast<const uint4*>(BTHi + boff);
    bh.u[1]  = *reinterpret_cast<const uint4*>(BTHi + boff + 16);
    bl.u[0]  = *reinterpret_cast<const uint4*>(BTLo + boff);
    bl.u[1]  = *reinterpret_cast<const uint4*>(BTLo + boff + 16);
    ah0.u[0] = *reinterpret_cast<const uint4*>(&tHi[lm][half * 8]);
    ah0.u[1] = *reinterpret_cast<const uint4*>(&tHi[lm][16 + half * 8]);
    al0.u[0] = *reinterpret_cast<const uint4*>(&tLo[lm][half * 8]);
    al0.u[1] = *reinterpret_cast<const uint4*>(&tLo[lm][16 + half * 8]);
    ah1.u[0] = *reinterpret_cast<const uint4*>(&tHi[16 + lm][half * 8]);
    ah1.u[1] = *reinterpret_cast<const uint4*>(&tHi[16 + lm][16 + half * 8]);
    al1.u[0] = *reinterpret_cast<const uint4*>(&tLo[16 + lm][half * 8]);
    al1.u[1] = *reinterpret_cast<const uint4*>(&tLo[16 + lm][16 + half * 8]);
    acc0 = WMMA_BF16(ah0.v, bh.v, acc0);
    acc1 = WMMA_BF16(ah1.v, bh.v, acc1);
    acc0 = WMMA_BF16(ah0.v, bl.v, acc0);
    acc1 = WMMA_BF16(ah1.v, bl.v, acc1);
    acc0 = WMMA_BF16(al0.v, bh.v, acc0);
    acc1 = WMMA_BF16(al1.v, bh.v, acc1);
}

// ---------------------------------------------------------------------------
// Fused GEMM (+ epilogue).  A (split bf16 [M][K]) @ W (split bf16 [N][K]^T).
//   mode 0:  X[m][n]     = acc + bvec[n]
//   mode 1:  f           = sigmoid(X + acc)
//            h_new[m][n] = (h + dt*f*A) / (1 + dt*(1/tau + f)), stored split.
// Block: 256 threads = 8 waves; macro-tile 32(M) x 128(N); per wave 2 C tiles.
// ---------------------------------------------------------------------------
__global__ __launch_bounds__(256) void gemm_ltc(
    const __bf16* __restrict__ AHi,   // [DIM][DIM] split-bf16 input (h or x)
    const __bf16* __restrict__ ALo,
    const __bf16* __restrict__ BTHi,  // [N][K] transposed weights, split
    const __bf16* __restrict__ BTLo,
    float*        __restrict__ X,     // mode0: write; mode1: read
    const float*  __restrict__ bvec,
    const float*  __restrict__ Avec,
    const float*  __restrict__ tauv,
    __bf16*       __restrict__ HoHi,  // mode1 outputs (split)
    __bf16*       __restrict__ HoLo,
    int mode)
{
    const int tid  = threadIdx.x;
    const int wave = tid >> 5;
    const int lane = tid & 31;
    const int half = lane >> 4;
    const int lm   = lane & 15;
    const int m0   = blockIdx.y * 32;
    const int n0   = blockIdx.x * 128 + wave * 16;

    // Double-buffered A tile: 32 rows x 32 k, hi + lo (8 KB total).
    __shared__ __align__(16) __bf16 sHi[2][32][32];
    __shared__ __align__(16) __bf16 sLo[2][32][32];

    // Staging map: 256 chunks of 16B (128 hi + 128 lo), one per thread.
    const int sel = tid >> 7;            // 0 = hi, 1 = lo
    const int c   = tid & 127;
    const int sm  = c >> 2;              // row 0..31
    const int sk  = (c & 3) * 8;         // k: 0,8,16,24

    const __bf16* __restrict__ Asrc = sel ? ALo : AHi;
    const size_t arow  = (size_t)(m0 + sm) * DIM + sk;
    __bf16* dst0 = sel ? &sLo[0][sm][sk] : &sHi[0][sm][sk];
    __bf16* dst1 = sel ? &sLo[1][sm][sk] : &sHi[1][sm][sk];
    const size_t bbase = (size_t)(n0 + lm) * DIM + half * 8;

    v8f acc0 = {}, acc1 = {};

    STAGE16(Asrc + arow, dst0);          // prologue: tile 0 -> buf0

    for (int it = 0; it < 32; it += 2) {
        STAGE16(Asrc + arow + (size_t)(it + 1) * 32, dst1);  // tile it+1
        AWAIT1();                        // tile it landed
        __syncthreads();
        mma_step(sHi[0], sLo[0], BTHi, BTLo, bbase + (size_t)it * 32,
                 lm, half, acc0, acc1);
        __syncthreads();                 // buf0 free for reuse

        if (it + 2 < 32) {
            STAGE16(Asrc + arow + (size_t)(it + 2) * 32, dst0);
            AWAIT1();                    // tile it+1 landed
        } else {
            AWAIT0();
        }
        __syncthreads();
        mma_step(sHi[1], sLo[1], BTHi, BTLo, bbase + (size_t)(it + 1) * 32,
                 lm, half, acc0, acc1);
        __syncthreads();                 // buf1 free for reuse
    }

    // C/D layout: element i -> row (tile_m + i + 8*half), col n0 + (lane&15).
    const int n = n0 + lm;
    if (mode == 0) {
        const float bn = bvec[n];
#pragma unroll
        for (int mt = 0; mt < 2; ++mt) {
            const v8f a = mt ? acc1 : acc0;
#pragma unroll
            for (int i = 0; i < 8; ++i) {
                const int m = m0 + mt * 16 + i + 8 * half;
                X[(size_t)m * DIM + n] = a[i] + bn;
            }
        }
    } else {
        const float An   = Avec[n];
        const float itau = 1.0f / tauv[n];
#pragma unroll
        for (int mt = 0; mt < 2; ++mt) {
            const v8f a = mt ? acc1 : acc0;
#pragma unroll
            for (int i = 0; i < 8; ++i) {
                const int m = m0 + mt * 16 + i + 8 * half;
                const size_t idx = (size_t)m * DIM + n;
                const float f  = 1.0f / (1.0f + __expf(-(X[idx] + a[i])));
                const float hv = (float)AHi[idx] + (float)ALo[idx];
                const float hn = (hv + DT * f * An) / (1.0f + DT * (itau + f));
                __bf16 hi, lo;
                split_bf16(hn, hi, lo);
                HoHi[idx] = hi;
                HoLo[idx] = lo;
            }
        }
    }
}

// ---------------------------------------------------------------------------
__global__ __launch_bounds__(256) void init_state(
    float* __restrict__ tot_h, float* __restrict__ halt_acc,
    float* __restrict__ tot_rem, float* __restrict__ tot_steps,
    int* __restrict__ cont)
{
    const size_t i = (size_t)blockIdx.x * 256 + threadIdx.x;
    tot_h[i] = 0.0f;
    if (i < DIM) {
        halt_acc[i]  = 0.0f;
        tot_rem[i]   = 0.0f;
        tot_steps[i] = 0.0f;
        cont[i]      = 1;
    }
}

// One ACT step: halting GEMV + state update + tot_h accumulation fused.
__global__ __launch_bounds__(256) void halt_step(
    const __bf16* __restrict__ hHi, const __bf16* __restrict__ hLo,
    const float* __restrict__ w_halt, const float* __restrict__ b_halt,
    float* __restrict__ tot_h, float* __restrict__ halt_acc,
    int* __restrict__ cont, float* __restrict__ tot_rem,
    float* __restrict__ tot_steps)
{
    const int r = blockIdx.x, tid = threadIdx.x;
    const size_t rb = (size_t)r * DIM;

    float p = 0.0f;
    for (int s = tid; s < DIM; s += 256)
        p += ((float)hHi[rb + s] + (float)hLo[rb + s]) * w_halt[s];
    __shared__ float red[256];
    red[tid] = p;
    __syncthreads();
    for (int o = 128; o > 0; o >>= 1) {
        if (tid < o) red[tid] += red[tid + o];
        __syncthreads();
    }

    __shared__ float coef_s;
    if (tid == 0) {
        const float sh = 1.0f / (1.0f + __expf(-(red[0] + b_halt[0])));
        const int   cc = cont[r];
        const float mh = (cc ? 1.0f : 0.0f) * sh;
        const float ha = halt_acc[r] + mh;
        const int ending = cc && (ha + sh > BUDGET);
        const int ncont  = cc && !ending;
        tot_steps[r] += ncont ? 1.0f : 0.0f;
        const float mr = ending ? (1.0f - ha) : 0.0f;
        tot_rem[r]  += mh;
        halt_acc[r]  = ha;
        cont[r]      = ncont;
        coef_s       = mh + mr;
    }
    __syncthreads();
    const float coef = coef_s;
    for (int s = tid; s < DIM; s += 256)
        tot_h[rb + s] += coef * ((float)hHi[rb + s] + (float)hLo[rb + s]);
}

// tot_h += cont * (1 - halt_acc) * h   (for-else remainder)
__global__ __launch_bounds__(256) void final_update(
    const __bf16* __restrict__ hHi, const __bf16* __restrict__ hLo,
    float* __restrict__ tot_h, const float* __restrict__ halt_acc,
    const int* __restrict__ cont)
{
    const int r = blockIdx.x, tid = threadIdx.x;
    const size_t rb = (size_t)r * DIM;
    const float coef = cont[r] ? (1.0f - halt_acc[r]) : 0.0f;
    for (int s = tid; s < DIM; s += 256)
        tot_h[rb + s] += coef * ((float)hHi[rb + s] + (float)hLo[rb + s]);
}

__global__ __launch_bounds__(1024) void finalize(
    const float* __restrict__ tot_rem, const float* __restrict__ tot_steps,
    float* __restrict__ out_tail)
{
    const int t = threadIdx.x;
    __shared__ float red[1024];
    red[t] = tot_rem[t];
    __syncthreads();
    for (int o = 512; o > 0; o >>= 1) {
        if (t < o) red[t] += red[t + o];
        __syncthreads();
    }
    if (t == 0) out_tail[0] = -TIME_PENALTY * (red[0] / (float)DIM);
    out_tail[1 + t] = tot_steps[t] + 1.0f;
}

// ---------------------------------------------------------------------------
extern "C" void kernel_launch(void* const* d_in, const int* in_sizes, int n_in,
                              void* d_out, int out_size, void* d_ws, size_t ws_size,
                              hipStream_t stream)
{
    const float* inputs  = (const float*)d_in[0];
    const float* hidden0 = (const float*)d_in[1];
    const float* Wx      = (const float*)d_in[2];
    const float* Wh      = (const float*)d_in[3];
    const float* bvec    = (const float*)d_in[4];
    const float* Avec    = (const float*)d_in[5];
    const float* tauv    = (const float*)d_in[6];
    const float* w_halt  = (const float*)d_in[7];
    const float* b_halt  = (const float*)d_in[8];
    float* out = (float*)d_out;

    const size_t MATB = (size_t)DIM * DIM;
    char* ws = (char*)d_ws;
    __bf16* WxT_hi = (__bf16*)ws; ws += MATB * 2;
    __bf16* WxT_lo = (__bf16*)ws; ws += MATB * 2;
    __bf16* WhT_hi = (__bf16*)ws; ws += MATB * 2;
    __bf16* WhT_lo = (__bf16*)ws; ws += MATB * 2;
    float*  X      = (float*)ws;  ws += MATB * 4;
    __bf16* inHi   = (__bf16*)ws; ws += MATB * 2;
    __bf16* inLo   = (__bf16*)ws; ws += MATB * 2;
    __bf16* h0Hi   = (__bf16*)ws; ws += MATB * 2;
    __bf16* h0Lo   = (__bf16*)ws; ws += MATB * 2;
    __bf16* h1Hi   = (__bf16*)ws; ws += MATB * 2;
    __bf16* h1Lo   = (__bf16*)ws; ws += MATB * 2;
    float*  halt_acc  = (float*)ws; ws += DIM * 4;
    float*  tot_rem   = (float*)ws; ws += DIM * 4;
    float*  tot_steps = (float*)ws; ws += DIM * 4;
    int*    cont      = (int*)ws;   ws += DIM * 4;

    float* tot_h = out;

    init_state<<<MATB / 256, 256, 0, stream>>>(tot_h, halt_acc, tot_rem,
                                               tot_steps, cont);

    dim3 tg(DIM / 32, DIM / 32), tb(32, 8);
    transpose_split_bf16<<<tg, tb, 0, stream>>>(Wx, WxT_hi, WxT_lo);
    transpose_split_bf16<<<tg, tb, 0, stream>>>(Wh, WhT_hi, WhT_lo);
    convert_split<<<MATB / 256, 256, 0, stream>>>(inputs, inHi, inLo);
    convert_split<<<MATB / 256, 256, 0, stream>>>(hidden0, h0Hi, h0Lo);

    dim3 gg(DIM / 128, DIM / 32), gb(256);
    // X = inputs @ Wx + b  (hoisted; invariant across steps/unfolds)
    gemm_ltc<<<gg, gb, 0, stream>>>(inHi, inLo, WxT_hi, WxT_lo, X, bvec,
                                    nullptr, nullptr, nullptr, nullptr, 0);

    __bf16 *hiIn = h0Hi, *loIn = h0Lo, *hiOut = h1Hi, *loOut = h1Lo;
    for (int t = 0; t < NSTEP; ++t) {
        for (int u = 0; u < NUNF; ++u) {
            gemm_ltc<<<gg, gb, 0, stream>>>(hiIn, loIn, WhT_hi, WhT_lo, X,
                                            nullptr, Avec, tauv, hiOut, loOut, 1);
            __bf16* th = hiIn; hiIn = hiOut; hiOut = th;
            __bf16* tl = loIn; loIn = loOut; loOut = tl;
        }
        halt_step<<<DIM, 256, 0, stream>>>(hiIn, loIn, w_halt, b_halt, tot_h,
                                           halt_acc, cont, tot_rem, tot_steps);
    }
    final_update<<<DIM, 256, 0, stream>>>(hiIn, loIn, tot_h, halt_acc, cont);
    finalize<<<1, 1024, 0, stream>>>(tot_rem, tot_steps, out + MATB);
}
